// LSTMTrajectoryPredictor_62466004353222
// MI455X (gfx1250) — matrix operand order repaired and checked by reading, not compiled, not run on hardware
//
#include <hip/hip_runtime.h>
#include <hip/hip_bf16.h>
#include <stdint.h>

// ---------------- problem constants ----------------
#define HSZ    512            // hidden size
#define GATES  2048           // 4*H
#define T_HIST 50
#define IN_SZ  5
#define OUT_SZ 2

// ---------------- WMMA types (CDNA5, wave32) ----------------
typedef __attribute__((ext_vector_type(16))) __bf16 v16bf;
typedef __attribute__((ext_vector_type(8)))  float  v8f;

union FragBF { v16bf v; uint32_t u[8]; };

// ---------------- tiling ----------------
#define BM 64                 // batch rows per block
#define BN 128                // permuted gate cols per block
#define KC 32                 // k-chunk (WMMA K)
#define NCH 16                // HSZ / KC chunks per GEMM pass
#define LSTRIDE 40            // bf16 elems per LDS row (32 + 8 pad = 80B, 16B-multiple)
#define ZSTRIDE 132           // fp32 z stage row stride

__device__ __forceinline__ float sigmoidf_(float x) { return 1.0f / (1.0f + __expf(-x)); }

// CDNA5 async memory->LDS DMA (ASYNCcnt-tracked, no VGPR round trip)
__device__ __forceinline__ void async_b128(uint32_t lds, const __bf16* g) {
    asm volatile("global_load_async_to_lds_b128 %0, %1, off"
                 :: "v"(lds), "v"((uint64_t)(uintptr_t)g) : "memory");
}

// Fused LSTM cell: z = hA @ Wh^T (+ xA @ Wx^T when NPASS==2) (+ small-K x term) + bias;
// gates; c,h update.  Wh/Wx are gate-permuted bf16 [2048][512] (row n' = 4*j + gate).
// Fully unrolled 2-deep async LDS pipeline (compile-time chunk count).
template <int NPASS>
__global__ __launch_bounds__(256)
void k_lstm_cell(const __bf16* __restrict__ hA, const __bf16* __restrict__ Wh,
                 const __bf16* __restrict__ xA, const __bf16* __restrict__ Wx,
                 const float*  __restrict__ xs, int xstride, int Kx,
                 const float*  __restrict__ Wxs,
                 const float*  __restrict__ bias,
                 float* __restrict__ c, __bf16* __restrict__ hOut)
{
    __shared__ __align__(16) __bf16 As[2][BM * LSTRIDE];
    __shared__ __align__(16) __bf16 Bs[2][BN * LSTRIDE];
    __shared__ __align__(16) float  zs[BM * ZSTRIDE];

    const int tid    = threadIdx.x;
    const int lane   = tid & 31;
    const int wid    = tid >> 5;
    const int wave_m = wid & 1;     // 2 wave-rows  * 32 rows
    const int wave_n = wid >> 1;    // 4 wave-cols  * 32 cols
    const int bm     = blockIdx.y * BM;
    const int bn     = blockIdx.x * BN;

    // per-thread staging coordinates: 64(A)+128(B) rows x 4 b128 segments
    const int srow = tid >> 2;      // 0..63
    const int sseg = tid & 3;       // 0..3 (8 bf16 each)

    // LDS destination byte offsets (wave-relative; low 32 bits of generic ptr)
    const uint32_t ldsA[2] = {
        (uint32_t)(uintptr_t)&As[0][srow * LSTRIDE + sseg * 8],
        (uint32_t)(uintptr_t)&As[1][srow * LSTRIDE + sseg * 8] };
    const uint32_t ldsB0[2] = {
        (uint32_t)(uintptr_t)&Bs[0][srow * LSTRIDE + sseg * 8],
        (uint32_t)(uintptr_t)&Bs[1][srow * LSTRIDE + sseg * 8] };
    const uint32_t ldsB1[2] = {
        (uint32_t)(uintptr_t)&Bs[0][(srow + 64) * LSTRIDE + sseg * 8],
        (uint32_t)(uintptr_t)&Bs[1][(srow + 64) * LSTRIDE + sseg * 8] };

    // per-pass global base addresses (pass index folds after unroll)
    const size_t rowA  = (size_t)(bm + srow) * HSZ + sseg * 8;
    const size_t rowB0 = (size_t)(bn + srow) * HSZ + sseg * 8;
    const size_t rowB1 = (size_t)(bn + srow + 64) * HSZ + sseg * 8;
    const __bf16* pA[2] = { hA, (NPASS == 2) ? xA : hA };
    const __bf16* pW[2] = { Wh, (NPASS == 2) ? Wx : Wh };

    v8f acc[2][2] = {};
    constexpr int nch = NPASS * NCH;

    // issue 3 async b128 loads (A tile row + 2 B tile rows) for chunk ch into parity par
    auto stage = [&](int ch, int par) {
        const int p  = ch >> 4;
        const int k0 = (ch & (NCH - 1)) * KC;
        async_b128(ldsA[par],  pA[p] + rowA  + k0);
        async_b128(ldsB0[par], pW[p] + rowB0 + k0);
        async_b128(ldsB1[par], pW[p] + rowB1 + k0);
    };

    stage(0, 0);                                // prologue: fill pipe

    #pragma unroll
    for (int ch = 0; ch < nch; ++ch) {
        const int par = ch & 1;
        if (ch + 1 < nch) {                     // folds at compile time after unroll
            stage(ch + 1, par ^ 1);             // prefetch next chunk into other buffer
            // async loads complete in order: <=3 outstanding => chunk ch's 3 landed
            asm volatile("s_wait_asynccnt 0x3" ::: "memory");
        } else {
            asm volatile("s_wait_asynccnt 0x0" ::: "memory");
        }
        __syncthreads();                        // all waves' DMA for chunk ch visible

        // fragments per ISA 16-bit A(16x32)/B(32x16) layouts
        FragBF a0, a1, b0, b1;
        {
            const int kap = (lane < 16) ? 0 : 4;  // A: k-halves {0..7,16..23} vs {8..15,24..31} (b32 pairs)
            const uint32_t* r0 = (const uint32_t*)(As[par] + (32 * wave_m +      (lane & 15)) * LSTRIDE);
            const uint32_t* r1 = (const uint32_t*)(As[par] + (32 * wave_m + 16 + (lane & 15)) * LSTRIDE);
            #pragma unroll
            for (int v = 0; v < 4; ++v) {
                a0.u[v] = r0[kap + v];  a0.u[v + 4] = r0[kap + 8 + v];
                a1.u[v] = r1[kap + v];  a1.u[v + 4] = r1[kap + 8 + v];
            }
            const int kbp = (lane < 16) ? 0 : 4;  // B: k-halves {0..7,16..23} vs {8..15,24..31} (b32 pairs)
            const uint32_t* c0p = (const uint32_t*)(Bs[par] + (32 * wave_n +      (lane & 15)) * LSTRIDE);
            const uint32_t* c1p = (const uint32_t*)(Bs[par] + (32 * wave_n + 16 + (lane & 15)) * LSTRIDE);
            #pragma unroll
            for (int v = 0; v < 8; ++v) { b0.u[v] = c0p[kbp + v + (v & 4)]; b1.u[v] = c1p[kbp + v + (v & 4)]; }
        }

        acc[0][0] = __builtin_amdgcn_wmma_f32_16x16x32_bf16(false, a0.v, false, b0.v, (short)0, acc[0][0], false, false);
        acc[0][1] = __builtin_amdgcn_wmma_f32_16x16x32_bf16(false, a0.v, false, b1.v, (short)0, acc[0][1], false, false);
        acc[1][0] = __builtin_amdgcn_wmma_f32_16x16x32_bf16(false, a1.v, false, b0.v, (short)0, acc[1][0], false, false);
        acc[1][1] = __builtin_amdgcn_wmma_f32_16x16x32_bf16(false, a1.v, false, b1.v, (short)0, acc[1][1], false, false);

        __syncthreads();                        // frag reads done before buffer reuse
    }

    // stage z tile to LDS (C layout: lanes 0-15 -> M=r, lanes 16-31 -> M=r+8; N = lane%16)
    #pragma unroll
    for (int tm = 0; tm < 2; ++tm)
        #pragma unroll
        for (int tn = 0; tn < 2; ++tn)
            #pragma unroll
            for (int r = 0; r < 8; ++r) {
                int m = 32 * wave_m + 16 * tm + ((lane < 16) ? r : (r + 8));
                int n = 32 * wave_n + 16 * tn + (lane & 15);
                zs[m * ZSTRIDE + n] = acc[tm][tn][r];
            }
    __syncthreads();

    // gate pass: cols bn..bn+127 cover hidden idx j = bn/4 .. bn/4+31, all 4 gates
    const int jl = tid & 31;
    const int rb = tid >> 5;
    const int jg = (bn >> 2) + jl;

    // hoisted loop-invariants: per-gate bias and tiny input-weight rows
    float bi4[4];
    #pragma unroll
    for (int g = 0; g < 4; ++g) bi4[g] = bias[bn + 4 * jl + g];
    float wx4[4][IN_SZ];
    if (Kx > 0) {
        #pragma unroll
        for (int g = 0; g < 4; ++g)
            for (int k = 0; k < Kx; ++k)
                wx4[g][k] = Wxs[(size_t)(bn + 4 * jl + g) * Kx + k];
    }

    #pragma unroll
    for (int rr = 0; rr < 8; ++rr) {
        int r = rb + rr * 8;
        int b = bm + r;
        float zi = zs[r * ZSTRIDE + 4 * jl + 0] + bi4[0];
        float zf = zs[r * ZSTRIDE + 4 * jl + 1] + bi4[1];
        float zg = zs[r * ZSTRIDE + 4 * jl + 2] + bi4[2];
        float zo = zs[r * ZSTRIDE + 4 * jl + 3] + bi4[3];
        if (Kx > 0) {  // tiny input matmul, scalar fp32
            const float* xr = xs + (size_t)b * xstride;
            for (int k = 0; k < Kx; ++k) {
                float xv = xr[k];
                zi += xv * wx4[0][k];
                zf += xv * wx4[1][k];
                zg += xv * wx4[2][k];
                zo += xv * wx4[3][k];
            }
        }
        float ig = sigmoidf_(zi), fg = sigmoidf_(zf), gg = tanhf(zg), og = sigmoidf_(zo);
        size_t ci = (size_t)b * HSZ + jg;
        float cn = fg * c[ci] + ig * gg;
        c[ci] = cn;
        hOut[ci] = (__bf16)(og * tanhf(cn));
    }
}

// pred = h1 @ outW^T + outb ; write to d_out[b][t][o] and feed back
__global__ __launch_bounds__(256)
void k_head(const __bf16* __restrict__ h1, const float* __restrict__ Wo,
            const float* __restrict__ bo, float* __restrict__ pred,
            float* __restrict__ out, int t, int Tf, int n)
{
    int idx = blockIdx.x * 256 + threadIdx.x;
    if (idx >= n) return;
    int b = idx >> 1, o = idx & 1;
    const __bf16* hr = h1 + (size_t)b * HSZ;
    const float*  wr = Wo + (size_t)o * HSZ;
    float acc = bo[o];
    for (int k = 0; k < HSZ; k += 8) {
        #pragma unroll
        for (int kk = 0; kk < 8; ++kk) acc += (float)hr[k + kk] * wr[k + kk];
    }
    pred[(size_t)b * 2 + o] = acc;
    out[(size_t)b * (Tf * 2) + t * 2 + o] = acc;
}

// ---------------- prep kernels ----------------
__global__ void k_permute_big(const float* __restrict__ W, __bf16* __restrict__ Wp, int n) {
    int idx = blockIdx.x * 256 + threadIdx.x;
    if (idx >= n) return;
    int np = idx >> 9, k = idx & 511;
    int j = np >> 2, g = np & 3;
    Wp[idx] = (__bf16)W[(size_t)(g * HSZ + j) * HSZ + k];
}
__global__ void k_bias(const float* __restrict__ bi, const float* __restrict__ bh, float* __restrict__ bp) {
    int np = blockIdx.x * 256 + threadIdx.x;
    if (np >= GATES) return;
    int j = np >> 2, g = np & 3, nsrc = g * HSZ + j;
    bp[np] = bi[nsrc] + bh[nsrc];
}
__global__ void k_permute_small(const float* __restrict__ W, float* __restrict__ Wp, int Kx, int n) {
    int idx = blockIdx.x * 256 + threadIdx.x;
    if (idx >= n) return;
    int np = idx / Kx, k = idx % Kx;
    int j = np >> 2, g = np & 3;
    Wp[idx] = W[(size_t)(g * HSZ + j) * Kx + k];
}
__global__ void k_zero(uint32_t* __restrict__ p, int n) {
    int i = blockIdx.x * 256 + threadIdx.x;
    if (i < n) p[i] = 0u;
}

// ---------------- host orchestration ----------------
extern "C" void kernel_launch(void* const* d_in, const int* in_sizes, int n_in,
                              void* d_out, int out_size, void* d_ws, size_t ws_size,
                              hipStream_t stream)
{
    const float* history  = (const float*)d_in[0];
    // d_in[1] = future_frames (device scalar; length derived from out_size instead)
    const float* encWih0 = (const float*)d_in[2];
    const float* encWhh0 = (const float*)d_in[3];
    const float* encbih0 = (const float*)d_in[4];
    const float* encbhh0 = (const float*)d_in[5];
    const float* encWih1 = (const float*)d_in[6];
    const float* encWhh1 = (const float*)d_in[7];
    const float* encbih1 = (const float*)d_in[8];
    const float* encbhh1 = (const float*)d_in[9];
    const float* decWih0 = (const float*)d_in[10];
    const float* decWhh0 = (const float*)d_in[11];
    const float* decbih0 = (const float*)d_in[12];
    const float* decbhh0 = (const float*)d_in[13];
    const float* decWih1 = (const float*)d_in[14];
    const float* decWhh1 = (const float*)d_in[15];
    const float* decbih1 = (const float*)d_in[16];
    const float* decbhh1 = (const float*)d_in[17];
    const float* outW    = (const float*)d_in[18];
    const float* outb    = (const float*)d_in[19];
    float* out = (float*)d_out;

    const int Bv = in_sizes[0] / (T_HIST * IN_SZ);        // 2048
    const int Tf = out_size / (Bv * OUT_SZ);              // 30

    // ---- workspace carve-out (256B aligned) ----
    char* ws = (char*)d_ws; size_t off = 0;
    auto take = [&](size_t bytes) -> char* {
        char* p = ws + off; off += (bytes + 255) & ~(size_t)255; return p;
    };
    const size_t bigW = (size_t)GATES * HSZ;              // 1M elems
    __bf16* pEncWhh0 = (__bf16*)take(bigW * 2);
    __bf16* pEncWih1 = (__bf16*)take(bigW * 2);
    __bf16* pEncWhh1 = (__bf16*)take(bigW * 2);
    __bf16* pDecWhh0 = (__bf16*)take(bigW * 2);
    __bf16* pDecWih1 = (__bf16*)take(bigW * 2);
    __bf16* pDecWhh1 = (__bf16*)take(bigW * 2);
    float* bEnc0 = (float*)take(GATES * 4);
    float* bEnc1 = (float*)take(GATES * 4);
    float* bDec0 = (float*)take(GATES * 4);
    float* bDec1 = (float*)take(GATES * 4);
    float* wxsEnc = (float*)take((size_t)GATES * IN_SZ * 4);
    float* wxsDec = (float*)take((size_t)GATES * OUT_SZ * 4);
    const size_t stateE = (size_t)Bv * HSZ;
    __bf16* h0[2] = { (__bf16*)take(stateE * 2), (__bf16*)take(stateE * 2) };
    __bf16* h1[2] = { (__bf16*)take(stateE * 2), (__bf16*)take(stateE * 2) };
    float* c0 = (float*)take(stateE * 4);
    float* c1 = (float*)take(stateE * 4);
    float* pred = (float*)take((size_t)Bv * OUT_SZ * 4);
    (void)ws_size; (void)n_in;

    // ---- prep: permute+convert weights, combine biases, zero state ----
    {
        int n = (int)bigW, blks = (n + 255) / 256;
        k_permute_big<<<blks, 256, 0, stream>>>(encWhh0, pEncWhh0, n);
        k_permute_big<<<blks, 256, 0, stream>>>(encWih1, pEncWih1, n);
        k_permute_big<<<blks, 256, 0, stream>>>(encWhh1, pEncWhh1, n);
        k_permute_big<<<blks, 256, 0, stream>>>(decWhh0, pDecWhh0, n);
        k_permute_big<<<blks, 256, 0, stream>>>(decWih1, pDecWih1, n);
        k_permute_big<<<blks, 256, 0, stream>>>(decWhh1, pDecWhh1, n);
        k_bias<<<(GATES + 255) / 256, 256, 0, stream>>>(encbih0, encbhh0, bEnc0);
        k_bias<<<(GATES + 255) / 256, 256, 0, stream>>>(encbih1, encbhh1, bEnc1);
        k_bias<<<(GATES + 255) / 256, 256, 0, stream>>>(decbih0, decbhh0, bDec0);
        k_bias<<<(GATES + 255) / 256, 256, 0, stream>>>(decbih1, decbhh1, bDec1);
        int ne = GATES * IN_SZ, nd = GATES * OUT_SZ;
        k_permute_small<<<(ne + 255) / 256, 256, 0, stream>>>(encWih0, wxsEnc, IN_SZ, ne);
        k_permute_small<<<(nd + 255) / 256, 256, 0, stream>>>(decWih0, wxsDec, OUT_SZ, nd);
        int hw = (int)(stateE / 2);   // bf16 buffer in u32 words
        int cw = (int)stateE;         // f32 buffer in u32 words
        k_zero<<<(hw + 255) / 256, 256, 0, stream>>>((uint32_t*)h0[0], hw);
        k_zero<<<(hw + 255) / 256, 256, 0, stream>>>((uint32_t*)h1[0], hw);
        k_zero<<<(cw + 255) / 256, 256, 0, stream>>>((uint32_t*)c0, cw);
        k_zero<<<(cw + 255) / 256, 256, 0, stream>>>((uint32_t*)c1, cw);
        int pw = Bv * OUT_SZ;
        k_zero<<<(pw + 255) / 256, 256, 0, stream>>>((uint32_t*)pred, pw);
    }

    const dim3 gridC(GATES / BN, Bv / BM);   // (16, 32)
    int p0 = 0, p1 = 0;

    // ---- encoder: 50 steps, 2 cells each ----
    for (int t = 0; t < T_HIST; ++t) {
        k_lstm_cell<1><<<gridC, 256, 0, stream>>>(
            h0[p0], pEncWhh0, (const __bf16*)nullptr, (const __bf16*)nullptr,
            history + (size_t)t * IN_SZ, T_HIST * IN_SZ, IN_SZ, wxsEnc,
            bEnc0, c0, h0[p0 ^ 1]);
        p0 ^= 1;
        k_lstm_cell<2><<<gridC, 256, 0, stream>>>(
            h1[p1], pEncWhh1, h0[p0], pEncWih1,
            (const float*)nullptr, 0, 0, (const float*)nullptr,
            bEnc1, c1, h1[p1 ^ 1]);
        p1 ^= 1;
    }

    // ---- decoder: Tf steps, 2 cells + head ----
    const int nhead = Bv * OUT_SZ;
    for (int t = 0; t < Tf; ++t) {
        k_lstm_cell<1><<<gridC, 256, 0, stream>>>(
            h0[p0], pDecWhh0, (const __bf16*)nullptr, (const __bf16*)nullptr,
            pred, OUT_SZ, OUT_SZ, wxsDec,
            bDec0, c0, h0[p0 ^ 1]);
        p0 ^= 1;
        k_lstm_cell<2><<<gridC, 256, 0, stream>>>(
            h1[p1], pDecWhh1, h0[p0], pDecWih1,
            (const float*)nullptr, 0, 0, (const float*)nullptr,
            bDec1, c1, h1[p1 ^ 1]);
        p1 ^= 1;
        k_head<<<(nhead + 255) / 256, 256, 0, stream>>>(
            h1[p1], outW, outb, pred, out, t, Tf, nhead);
    }
}
